// NEP_50405736186589
// MI455X (gfx1250) — matrix operand order repaired and checked by reading, not compiled, not run on hardware
//
#include <hip/hip_runtime.h>
#include <hip/hip_bf16.h>
#include <math.h>

typedef __attribute__((ext_vector_type(16))) _Float16 v16h;
typedef __attribute__((ext_vector_type(8)))  _Float16 v8h;
typedef __attribute__((ext_vector_type(8)))  float    v8f;
typedef __attribute__((ext_vector_type(2)))  float    v2f;

#define NATOMS  4096
#define NTYPES  2
#define MAXNEI  100
#define MTOT    (NTYPES*MAXNEI)   /* 200 */
#define NMAXP1  5
#define NBASEP1 9
#define CPAD    10                /* coeff row padded to 10 floats: v2f aligned */
#define NFEATS  35
#define NEURON  80
#define KPAD    64                /* K padded for WMMA (2 x 32) */
#define RC_R    8.0f
#define RC_A    4.0f
#define PI_F    3.14159265358979323846f

#if defined(__AMDGCN__) && __has_builtin(__builtin_amdgcn_tanhf)
#define FAST_TANH(x) __builtin_amdgcn_tanhf(x)
#else
#define FAST_TANH(x) tanhf(x)
#endif

// W3B = C3B * multiplicity (m=0 once, m>0 cos/sin pairs twice)
__device__ __constant__ float dW3B[24] = {
  0.238732414637843f, 2.f*0.119366207318922f, 2.f*0.119366207318922f,
  0.099471839432435f, 2.f*0.596831036594608f, 2.f*0.596831036594608f,
  2.f*0.149207759148652f, 2.f*0.149207759148652f,
  0.139260575205408f, 2.f*0.104445431404056f, 2.f*0.104445431404056f,
  2.f*1.044454314040563f, 2.f*1.044454314040563f,
  2.f*0.174075719006761f, 2.f*0.174075719006761f,
  0.011190581936149f, 2.f*0.223811638722978f, 2.f*0.223811638722978f,
  2.f*0.111905819361489f, 2.f*0.111905819361489f,
  2.f*1.566681471060845f, 2.f*1.566681471060845f,
  2.f*0.195835183882606f, 2.f*0.195835183882606f };

__device__ __constant__ float dC4B[5] = {
 -0.007499480826664f, -0.134990654879954f, 0.067495327439977f,
  0.404971964639861f, -0.809943929279723f };
__device__ __constant__ float dC5B[3] = {
  0.026596810706114f, 0.053193621412227f, 0.026596810706114f };

// ---------------------------------------------------------------------------
// VALU lane-xor swap within 16-lane halves via v_permlane16_b32 (avoids the
// LDS/DS pipe that ds_bpermute-based __shfl_xor uses).
// ---------------------------------------------------------------------------
__device__ __forceinline__ float permxor16(float x, int sel0, int sel1) {
  int xi = __float_as_int(x);
  int r  = __builtin_amdgcn_permlane16(xi, xi, sel0, sel1, false, false);
  return __int_as_float(r);
}
#define PERMXOR1(x)  permxor16((x), 0x67452301, 0xEFCDAB89)  /* lane ^ 1 */
#define PERMXOR2(x)  permxor16((x), 0x54761032, 0xDCFE98BA)  /* lane ^ 2 */
#define PERMXOR4(x)  permxor16((x), 0x32107654, 0xBA98FEDC)  /* lane ^ 4 */
#define PERMXOR8(x)  permxor16((x), 0xFEDCBA98, 0x76543210)  /* lane ^ 8 */

__device__ __forceinline__ float half16_reduce(float x) {
  x += PERMXOR1(x);
  x += PERMXOR2(x);
  x += PERMXOR4(x);
  x += PERMXOR8(x);
  return x;  // every lane holds the sum of its 16-lane half
}

// ---------------------------------------------------------------------------
// Pad + transpose W1 [2][35][80] f32 -> W1pT [2][80][64] f16 (zero-padded K)
// so each WMMA B-fragment is one contiguous 32-byte vector load.
// ---------------------------------------------------------------------------
__global__ __launch_bounds__(256) void nep_padw1(const float* __restrict__ W1,
                                                 _Float16* __restrict__ W1pT) {
  int idx = blockIdx.x * blockDim.x + threadIdx.x;
  const int total = NTYPES * NEURON * KPAD;
  if (idx >= total) return;
  int k = idx % KPAD;
  int h = (idx / KPAD) % NEURON;
  int t = idx / (KPAD * NEURON);
  W1pT[idx] = (k < NFEATS) ? (_Float16)W1[(t * NFEATS + k) * NEURON + h]
                           : (_Float16)0.0f;
}

// ---------------------------------------------------------------------------
// Descriptor stage: TWO ATOMS PER WAVE (16 lanes per atom; 2048 waves).
// Lanes stride over the 200 neighbors (coalesced float4 loads), accumulate
// per-lane partials with PACKED f32 FMAs (v_pk_fma_f32): s kept as float2
// pairs, basis contractions as 4x float2 + 1 scalar (LDS coeff rows padded to
// 10 floats for 8B alignment). A 4-step permlane16 xor tree reduces the 125
// partials within each half; lanes 0/16 assemble invariants, store f16 rows.
// ---------------------------------------------------------------------------
__global__ __launch_bounds__(256) void nep_descriptor(
    const float* __restrict__ ImageDR, const int* __restrict__ tmap,
    const float* __restrict__ c2g, const float* __restrict__ c3g,
    _Float16* __restrict__ featH) {
  const int NROWS = NTYPES * NTYPES * NMAXP1;          // 20 coeff rows
  __shared__ float c2s[NROWS * CPAD];
  __shared__ float c3s[NROWS * CPAD];
  for (int t = threadIdx.x; t < NROWS * NBASEP1; t += blockDim.x) {
    int k = t % NBASEP1, row = t / NBASEP1;
    c2s[row * CPAD + k] = c2g[t];
    c3s[row * CPAD + k] = c3g[t];
  }
  __syncthreads();

  const int lane = threadIdx.x & 31;
  const int sub  = lane & 15;            // lane within the atom's half
  const int half = lane >> 4;            // which atom of this wave
  const int atom = blockIdx.x * ((int)blockDim.x >> 4) +
                   ((threadIdx.x >> 5) << 1) + half;
  const int ti   = tmap[atom];

  float q2[NMAXP1] = {0.f, 0.f, 0.f, 0.f, 0.f};
  v2f s2[NMAXP1][12];
#pragma unroll
  for (int n = 0; n < NMAXP1; n++)
#pragma unroll
    for (int j = 0; j < 12; j++) s2[n][j] = (v2f){0.f, 0.f};

  const float* base = ImageDR + (size_t)atom * MTOT * 4;

#pragma unroll
  for (int it = 0; it < (MTOT + 15) / 16; it++) {       // 13 iterations
    const int m = it * 16 + sub;
    float4 d = make_float4(0.f, 0.f, 0.f, 0.f);
    if (m < MTOT) d = reinterpret_cast<const float4*>(base)[m];

    const float r     = d.x;
    const bool  valid = r > 1e-6f;
    const float inv   = valid ? (1.0f / r) : 0.0f;
    const float ux = d.y * inv, uy = d.z * inv, uz = d.w * inv;
    const int   jt = (m >= MAXNEI) ? 1 : 0;             // junk for m>=200: gn==0
    const float* c2b = &c2s[(ti * NTYPES + jt) * NMAXP1 * CPAD];
    const float* c3b = &c3s[(ti * NTYPES + jt) * NMAXP1 * CPAD];

    // ---- radial basis (rc = 8) -> q2 partial (packed dot products) ----
    {
      float f[NBASEP1];
      float fc = (valid && r < RC_R)
                     ? 0.5f * (__cosf(PI_F * r * (1.0f / RC_R)) + 1.0f) : 0.0f;
      float xx = r * (1.0f / RC_R) - 1.0f;
      float x  = 2.0f * xx * xx - 1.0f;
      f[0] = fc;
      f[1] = 0.5f * (x + 1.0f) * fc;
      float Tkm1 = 1.0f, Tk = x;
#pragma unroll
      for (int k = 2; k < NBASEP1; k++) {
        float Tkp1 = 2.0f * x * Tk - Tkm1;
        f[k] = 0.5f * (Tkp1 + 1.0f) * fc;
        Tkm1 = Tk; Tk = Tkp1;
      }
      v2f f2[4];
#pragma unroll
      for (int j = 0; j < 4; j++) f2[j] = (v2f){f[2 * j], f[2 * j + 1]};
#pragma unroll
      for (int n = 0; n < NMAXP1; n++) {
        v2f acc = (v2f){0.f, 0.f};
#pragma unroll
        for (int j = 0; j < 4; j++)
          acc = __builtin_elementwise_fma(f2[j],
                    *(const v2f*)&c2b[n * CPAD + 2 * j], acc);
        q2[n] += acc.x + acc.y + f[8] * c2b[n * CPAD + 8];
      }
    }

    // ---- angular basis (rc = 4) -> gn (packed dot products) ----
    float gn[NMAXP1];
    {
      float f[NBASEP1];
      float fc = (valid && r < RC_A)
                     ? 0.5f * (__cosf(PI_F * r * (1.0f / RC_A)) + 1.0f) : 0.0f;
      float xx = r * (1.0f / RC_A) - 1.0f;
      float x  = 2.0f * xx * xx - 1.0f;
      f[0] = fc;
      f[1] = 0.5f * (x + 1.0f) * fc;
      float Tkm1 = 1.0f, Tk = x;
#pragma unroll
      for (int k = 2; k < NBASEP1; k++) {
        float Tkp1 = 2.0f * x * Tk - Tkm1;
        f[k] = 0.5f * (Tkp1 + 1.0f) * fc;
        Tkm1 = Tk; Tk = Tkp1;
      }
      v2f f2[4];
#pragma unroll
      for (int j = 0; j < 4; j++) f2[j] = (v2f){f[2 * j], f[2 * j + 1]};
#pragma unroll
      for (int n = 0; n < NMAXP1; n++) {
        v2f acc = (v2f){0.f, 0.f};
#pragma unroll
        for (int j = 0; j < 4; j++)
          acc = __builtin_elementwise_fma(f2[j],
                    *(const v2f*)&c3b[n * CPAD + 2 * j], acc);
        gn[n] = acc.x + acc.y + f[8] * c3b[n * CPAD + 8];
      }
    }

    // ---- 24 solid-harmonic polynomials of u ----
    const float X = ux, Y = uy, Z = uz;
    const float X2 = X * X, Y2 = Y * Y, Z2 = Z * Z;
    float blm[24];
    blm[0] = Z; blm[1] = X; blm[2] = Y;
    blm[3] = 3.f * Z2 - 1.f; blm[4] = X * Z; blm[5] = Y * Z;
    blm[6] = X2 - Y2; blm[7] = 2.f * X * Y;
    blm[8]  = (5.f * Z2 - 3.f) * Z;
    blm[9]  = (5.f * Z2 - 1.f) * X;
    blm[10] = (5.f * Z2 - 1.f) * Y;
    blm[11] = (X2 - Y2) * Z;
    blm[12] = 2.f * X * Y * Z;
    blm[13] = (X2 - 3.f * Y2) * X;
    blm[14] = (3.f * X2 - Y2) * Y;
    blm[15] = (35.f * Z2 - 30.f) * Z2 + 3.f;
    blm[16] = (7.f * Z2 - 3.f) * X * Z;
    blm[17] = (7.f * Z2 - 3.f) * Y * Z;
    blm[18] = (7.f * Z2 - 1.f) * (X2 - Y2);
    blm[19] = (7.f * Z2 - 1.f) * 2.f * X * Y;
    blm[20] = (X2 - 3.f * Y2) * X * Z;
    blm[21] = (3.f * X2 - Y2) * Y * Z;
    blm[22] = (X2 - 3.f * Y2) * X2 - (3.f * X2 - Y2) * Y2;
    blm[23] = 4.f * (X2 - Y2) * X * Y;

    // ---- rank-1 update with packed FMAs: 60 v_pk_fma_f32 per neighbor ----
    v2f blm2[12];
#pragma unroll
    for (int j = 0; j < 12; j++) blm2[j] = (v2f){blm[2 * j], blm[2 * j + 1]};
#pragma unroll
    for (int n = 0; n < NMAXP1; n++) {
      v2f g2 = (v2f){gn[n], gn[n]};
#pragma unroll
      for (int j = 0; j < 12; j++)
        s2[n][j] = __builtin_elementwise_fma(g2, blm2[j], s2[n][j]);
    }
  }

  // ---- unpack, then per-half (16-lane) VALU tree reduction (125 values) ----
  float s[NMAXP1][24];
#pragma unroll
  for (int n = 0; n < NMAXP1; n++) {
    q2[n] = half16_reduce(q2[n]);
#pragma unroll
    for (int j = 0; j < 12; j++) {
      s[n][2 * j]     = half16_reduce(s2[n][j].x);
      s[n][2 * j + 1] = half16_reduce(s2[n][j].y);
    }
  }

  if (sub == 0) {   // lanes 0 and 16: one atom each
    _Float16 fv[KPAD];
#pragma unroll
    for (int n = 0; n < NMAXP1; n++) fv[n] = (_Float16)q2[n];

    const int LO[5] = {0, 3, 8, 15, 24};
#pragma unroll
    for (int b = 0; b < 4; b++) {
#pragma unroll
      for (int n = 0; n < NMAXP1; n++) {
        float acc = 0.f;
#pragma unroll 24
        for (int p = LO[b]; p < LO[b + 1]; p++) {
          float sv = s[n][p];
          acc = fmaf(dW3B[p], sv * sv, acc);
        }
        fv[5 + b * NMAXP1 + n] = (_Float16)acc;
      }
    }
#pragma unroll
    for (int n = 0; n < NMAXP1; n++) {
      float s8 = s[n][8], s9 = s[n][9], s10 = s[n][10], s11 = s[n][11], s12 = s[n][12];
      float q4 = dC4B[0] * s8 * s8 * s8
               + dC4B[1] * s8 * (s9 * s9 + s10 * s10)
               + dC4B[2] * s8 * (s11 * s11 + s12 * s12)
               + dC4B[3] * s11 * (s10 * s10 - s9 * s9)
               + dC4B[4] * s9 * s10 * s12;
      fv[25 + n] = (_Float16)q4;
      float s0 = s[n][0], s1 = s[n][1], s2v = s[n][2];
      float p12 = s1 * s1 + s2v * s2v;
      float q5 = dC5B[0] * s0 * s0 * s0 * s0
               + dC5B[1] * s0 * s0 * p12
               + dC5B[2] * p12 * p12;
      fv[30 + n] = (_Float16)q5;
    }
#pragma unroll
    for (int k = NFEATS; k < KPAD; k++) fv[k] = (_Float16)0.f;

    _Float16* fo = featH + (size_t)atom * KPAD;
#pragma unroll
    for (int k = 0; k < KPAD; k++) fo[k] = fv[k];
  }
}

// ---------------------------------------------------------------------------
// Fitting MLP with WMMA: one wave per 16-atom tile, f16 in / f32 accumulate.
// Fragments are contiguous vector loads (feat row-major f16, W1 transposed
// [type][neuron][Kpad] f16). Both type weights multiplied; per-row select by
// atom type, hardware tanh, 80->1 projection via permlane16 half-wave reduce.
// ---------------------------------------------------------------------------
__global__ __launch_bounds__(256) void nep_mlp_wmma(
    const _Float16* __restrict__ featH, const int* __restrict__ tmap,
    const _Float16* __restrict__ W1pT, const float* __restrict__ b1,
    const float* __restrict__ W2, const float* __restrict__ b2,
    float* __restrict__ Ei) {
  const int lane  = threadIdx.x & 31;
  const int tile  = blockIdx.x * (blockDim.x >> 5) + (threadIdx.x >> 5);
  const int mrow  = lane & 15;
  const int khalf = lane >> 4;
  const int ncol0 = lane & 15;

  // A fragments: one contiguous 16-half run per k-step (k = 16*khalf + s), matching B.
  const _Float16* frow = featH + (size_t)(tile * 16 + mrow) * KPAD;
  v16h a[2];
#pragma unroll
  for (int ks = 0; ks < 2; ks++) {
    a[ks] = *(const v16h*)(frow + ks * 32 + khalf * 16);
  }

  int ta[8];
#pragma unroll
  for (int v = 0; v < 8; v++) ta[v] = tmap[tile * 16 + v + 8 * khalf];

  float epart[8] = {0.f, 0.f, 0.f, 0.f, 0.f, 0.f, 0.f, 0.f};

#pragma unroll
  for (int nt = 0; nt < 5; nt++) {                // N = 80 -> 5 tiles of 16
    const int ncol = nt * 16 + ncol0;

    // B fragments: one contiguous 16-half run per (type, k-step).
    const _Float16* w0 = W1pT + (size_t)(0 * NEURON + ncol) * KPAD + khalf * 16;
    const _Float16* w1 = W1pT + (size_t)(1 * NEURON + ncol) * KPAD + khalf * 16;
    v16h b00 = *(const v16h*)(w0);
    v16h b01 = *(const v16h*)(w0 + 32);
    v16h b10 = *(const v16h*)(w1);
    v16h b11 = *(const v16h*)(w1 + 32);

    v8f c0 = {0.f, 0.f, 0.f, 0.f, 0.f, 0.f, 0.f, 0.f};
    v8f c1 = {0.f, 0.f, 0.f, 0.f, 0.f, 0.f, 0.f, 0.f};
    c0 = __builtin_amdgcn_wmma_f32_16x16x32_f16(false, a[0], false, b00,
                                                (short)0, c0, false, false);
    c0 = __builtin_amdgcn_wmma_f32_16x16x32_f16(false, a[1], false, b01,
                                                (short)0, c0, false, false);
    c1 = __builtin_amdgcn_wmma_f32_16x16x32_f16(false, a[0], false, b10,
                                                (short)0, c1, false, false);
    c1 = __builtin_amdgcn_wmma_f32_16x16x32_f16(false, a[1], false, b11,
                                                (short)0, c1, false, false);

    const float b1v0 = b1[0 * NEURON + ncol];
    const float b1v1 = b1[1 * NEURON + ncol];
    const float w2v0 = W2[0 * NEURON + ncol];
    const float w2v1 = W2[1 * NEURON + ncol];

    // C layout: lanes 0-15 hold rows M=vgpr, lanes 16-31 rows M=vgpr+8.
#pragma unroll
    for (int v = 0; v < 8; v++) {
      const int t = ta[v];
      float acc = t ? c1[v] : c0[v];
      float h = FAST_TANH(acc + (t ? b1v1 : b1v0));
      epart[v] = fmaf(h, t ? w2v1 : w2v0, epart[v]);
    }
  }

  // Reduce the 80 hidden contributions: 16 lanes of each half hold distinct n.
#pragma unroll
  for (int v = 0; v < 8; v++) epart[v] = half16_reduce(epart[v]);

  if (ncol0 == 0) {
#pragma unroll
    for (int v = 0; v < 8; v++) {
      int atom = tile * 16 + v + 8 * khalf;
      Ei[atom] = epart[v] + b2[ta[v]];
    }
  }
}

// ---------------------------------------------------------------------------
// Deterministic single-block reduction Ei -> Etot.
// ---------------------------------------------------------------------------
__global__ __launch_bounds__(256) void nep_reduce(const float* __restrict__ Ei,
                                                  float* __restrict__ Etot) {
  __shared__ float sm[256];
  float s = 0.f;
  for (int i = threadIdx.x; i < NATOMS; i += 256) s += Ei[i];
  sm[threadIdx.x] = s;
  __syncthreads();
  for (int off = 128; off > 0; off >>= 1) {
    if (threadIdx.x < off) sm[threadIdx.x] += sm[threadIdx.x + off];
    __syncthreads();
  }
  if (threadIdx.x == 0) Etot[0] = sm[0];
}

extern "C" void kernel_launch(void* const* d_in, const int* in_sizes, int n_in,
                              void* d_out, int out_size, void* d_ws, size_t ws_size,
                              hipStream_t stream) {
  (void)in_sizes; (void)n_in; (void)out_size; (void)ws_size;
  // setup_inputs order:
  // 0 list_neigh(i32)  1 Imagetype_map(i32)  2 atom_type(i32)  3 ImageDR(f32)
  // 4 nghost(i32)      5 c2(f32)             6 c3(f32)         7 W1(f32)
  // 8 b1(f32)          9 W2(f32)            10 b2(f32)
  const int*   tmap    = (const int*)d_in[1];
  const float* ImageDR = (const float*)d_in[3];
  const float* c2      = (const float*)d_in[5];
  const float* c3      = (const float*)d_in[6];
  const float* W1      = (const float*)d_in[7];
  const float* b1      = (const float*)d_in[8];
  const float* W2      = (const float*)d_in[9];
  const float* b2      = (const float*)d_in[10];

  float* out = (float*)d_out;                       // [0]=Etot, [1..4096]=Ei

  _Float16* featH = (_Float16*)d_ws;                // 4096*64 f16
  _Float16* W1pT  = featH + (size_t)NATOMS * KPAD;  // 2*80*64 f16

  nep_padw1<<<(NTYPES * NEURON * KPAD + 255) / 256, 256, 0, stream>>>(W1, W1pT);
  nep_descriptor<<<NATOMS / 16, 256, 0, stream>>>(ImageDR, tmap, c2, c3, featH);
  nep_mlp_wmma<<<(NATOMS / 16) / 8, 256, 0, stream>>>(featH, tmap, W1pT, b1, W2, b2,
                                                      out + 1);
  nep_reduce<<<1, 256, 0, stream>>>(out + 1, out);
}
